// AdditiveAttention_73839077753442
// MI455X (gfx1250) — matrix operand order repaired and checked by reading, not compiled, not run on hardware
//
#include <hip/hip_runtime.h>
#include <hip/hip_bf16.h>

#define BATCH 32
#define SEQ   2048
#define KD    512
#define QD    1024
#define HD    128

typedef __attribute__((ext_vector_type(16))) __bf16 v16bf;
typedef __attribute__((ext_vector_type(8)))  float  v8f;

// ---------------------------------------------------------------------------
// q_proj[b,h] = sum_q query[b,q] * W1[h,q]   (tiny: 8.4 MFLOP)
// ---------------------------------------------------------------------------
__global__ void qproj_kernel(const float* __restrict__ query,
                             const float* __restrict__ W1,
                             float* __restrict__ qproj) {
    int b = blockIdx.x;
    int h = threadIdx.x;                 // 128 threads
    const float* q = query + (size_t)b * QD;
    const float* w = W1    + (size_t)h * QD;
    float s = 0.f;
    for (int i = 0; i < QD; i += 4) {
        float4 qv = *(const float4*)(q + i);
        float4 wv = *(const float4*)(w + i);
        s += qv.x * wv.x + qv.y * wv.y + qv.z * wv.z + qv.w * wv.w;
    }
    qproj[b * HD + h] = s;
}

// ---------------------------------------------------------------------------
// Pre-swizzle W2 (f32, [H,K]) into bf16 B-operand lane layout:
//   w2bf[((t*16 + ks)*32 + lane)*16 + e] = bf16( W2[t*16 + (lane&15)][ks*32 + (lane>>4)*16 + e] )
// ---------------------------------------------------------------------------
__global__ void w2conv_kernel(const float* __restrict__ W2,
                              __bf16* __restrict__ w2bf) {
    int ts   = blockIdx.x;               // t*16 + ks, 0..127
    int lane = threadIdx.x;              // 32
    int t = ts >> 4, ks = ts & 15;
    int n = lane & 15, g = lane >> 4;
    const float* src = W2 + (size_t)(t * 16 + n) * KD + ks * 32 + g * 16;
    __bf16* dst = w2bf + ((size_t)(ts * 32) + lane) * 16;
    #pragma unroll
    for (int e = 0; e < 16; ++e) dst[e] = (__bf16)src[e];
}

// ---------------------------------------------------------------------------
// 8 waves/block; block stages swizzled W2 (128 KB bf16) into LDS once, then
// each wave computes a 16-row seq tile: 16 K-steps x 8 H-tiles = 128 wmma,
// with B software-pipelined 2 tiles deep from LDS and A double-buffered from
// global (f32 -> bf16 converted in registers).
// ---------------------------------------------------------------------------
__global__ void scores_kernel(const float* __restrict__ keys,
                              const float* __restrict__ qproj,
                              const float* __restrict__ Vv,
                              const __bf16* __restrict__ w2bf,
                              float* __restrict__ scores) {
    extern __shared__ __bf16 Blds[];     // 8*16*32*16 bf16 = 128 KB

    // Cooperative stage: global -> LDS (one shot, amortized over 1024 WMMAs).
    {
        const uint4* src = (const uint4*)w2bf;
        uint4*       dst = (uint4*)Blds;
        for (int i = threadIdx.x; i < (HD * KD) / 8; i += 256)   // 8192 uint4
            dst[i] = src[i];
    }
    __syncthreads();

    int b    = blockIdx.y;
    int wave = threadIdx.x >> 5;
    int lane = threadIdx.x & 31;
    int s0   = (blockIdx.x * 8 + wave) * 16;
    int n = lane & 15, g = lane >> 4;

    const float* arow = keys + ((size_t)b * SEQ + s0 + n) * KD;
    const int    o    = g * 16;
    const v16bf* Bl   = (const v16bf*)Blds;

    // A operand for K-step ks: this lane's row, K = ks*32 + {o..o+15}
    auto loadA = [&](int ks) {
        const float4* p = (const float4*)(arow + ks * 32 + o);
        float4 a0 = p[0], a1 = p[1], a2 = p[2], a3 = p[3];
        v16bf A;
        A[0]=(__bf16)a0.x; A[1]=(__bf16)a0.y; A[2]=(__bf16)a0.z; A[3]=(__bf16)a0.w;
        A[4]=(__bf16)a1.x; A[5]=(__bf16)a1.y; A[6]=(__bf16)a1.z; A[7]=(__bf16)a1.w;
        A[8]=(__bf16)a2.x; A[9]=(__bf16)a2.y; A[10]=(__bf16)a2.z; A[11]=(__bf16)a2.w;
        A[12]=(__bf16)a3.x; A[13]=(__bf16)a3.y; A[14]=(__bf16)a3.z; A[15]=(__bf16)a3.w;
        return A;
    };
    // Linear iteration i = ks*8 + t  ->  LDS v16bf index for (t, ks).
    auto bofs = [&](int i) {
        int t = i & 7, ks = i >> 3;
        return (t * 16 + ks) * 32 + lane;
    };

    v8f acc[8] = {};                     // 8 H-tiles of 16x16 f32

    // Pipeline prologue: A one K-step deep, B two tiles deep.
    v16bf Acur  = loadA(0);
    v16bf Bbuf0 = Bl[bofs(0)];
    v16bf Bbuf1 = Bl[bofs(1)];

    for (int ks = 0; ks < 16; ++ks) {
        v16bf Anext = loadA(ks < 15 ? ks + 1 : 15);   // branch-free; last iter reloads
        #pragma unroll
        for (int t = 0; t < 8; ++t) {
            acc[t] = __builtin_amdgcn_wmma_f32_16x16x32_bf16(
                /*neg_a=*/false, Acur, /*neg_b=*/false, (t & 1) ? Bbuf1 : Bbuf0,
                /*c_mod=*/(short)0, acc[t],
                /*reuse_a=*/false, /*reuse_b=*/false);
            // Refill this slot with tile i+2 (clamped, branch-free).
            int ni = ks * 8 + t + 2;
            ni = ni < 128 ? ni : 127;
            if (t & 1) Bbuf1 = Bl[bofs(ni)];
            else       Bbuf0 = Bl[bofs(ni)];
        }
        Acur = Anext;
    }

    // Epilogue: lane owns column n of every H-tile, rows m = r + 8*g.
    float part[8];
    #pragma unroll
    for (int r = 0; r < 8; ++r) part[r] = 0.f;
    #pragma unroll
    for (int t = 0; t < 8; ++t) {
        float qp = qproj[b * HD + t * 16 + n];
        float vv = Vv[t * 16 + n];
        #pragma unroll
        for (int r = 0; r < 8; ++r)
            part[r] += tanhf(qp + acc[t][r]) * vv;
    }
    // Reduce over the 16 n-lanes within each half-wave (masks preserve halves).
    #pragma unroll
    for (int m = 1; m < 16; m <<= 1)
        #pragma unroll
        for (int r = 0; r < 8; ++r)
            part[r] += __shfl_xor(part[r], m, 32);

    if (lane == 0 || lane == 16) {
        int mbase = g * 8;
        #pragma unroll
        for (int r = 0; r < 8; ++r)
            scores[(size_t)b * SEQ + s0 + mbase + r] = part[r];
    }
}

// ---------------------------------------------------------------------------
// Softmax over S per batch.
// ---------------------------------------------------------------------------
__global__ void softmax_kernel(const float* __restrict__ scores,
                               float* __restrict__ weights) {
    int b = blockIdx.x;
    __shared__ float red[256];
    const float* s = scores + (size_t)b * SEQ;
    float* w = weights + (size_t)b * SEQ;
    int tid = threadIdx.x;

    float mx = -__builtin_inff();
    for (int i = tid; i < SEQ; i += 256) mx = fmaxf(mx, s[i]);
    red[tid] = mx; __syncthreads();
    for (int k = 128; k > 0; k >>= 1) {
        if (tid < k) red[tid] = fmaxf(red[tid], red[tid + k]);
        __syncthreads();
    }
    mx = red[0]; __syncthreads();

    float sum = 0.f;
    for (int i = tid; i < SEQ; i += 256) {
        float e = __expf(s[i] - mx);
        w[i] = e;
        sum += e;
    }
    red[tid] = sum; __syncthreads();
    for (int k = 128; k > 0; k >>= 1) {
        if (tid < k) red[tid] += red[tid + k];
        __syncthreads();
    }
    float inv = 1.f / red[0];
    for (int i = tid; i < SEQ; i += 256) w[i] *= inv;
}

// ---------------------------------------------------------------------------
// context[b,k] = sum_s w[b,s] * keys[b,s,k]; keys should be L2-resident (134MB < 192MB).
// ---------------------------------------------------------------------------
__global__ void context_kernel(const float* __restrict__ keys,
                               const float* __restrict__ weights,
                               float* __restrict__ context) {
    int b = blockIdx.x;
    int k = threadIdx.x;                 // 512 threads
    __shared__ float w[SEQ];
    for (int i = threadIdx.x; i < SEQ; i += blockDim.x)
        w[i] = weights[(size_t)b * SEQ + i];
    __syncthreads();

    const float* kb = keys + (size_t)b * SEQ * KD + k;
    float acc = 0.f;
    for (int s = 0; s < SEQ; ++s)
        acc += w[s] * kb[(size_t)s * KD];
    context[b * KD + k] = acc;
}

// ---------------------------------------------------------------------------
extern "C" void kernel_launch(void* const* d_in, const int* in_sizes, int n_in,
                              void* d_out, int out_size, void* d_ws, size_t ws_size,
                              hipStream_t stream) {
    const float* query = (const float*)d_in[0];   // [32,1024]
    const float* keys  = (const float*)d_in[1];   // [32,2048,512]
    const float* W1    = (const float*)d_in[2];   // [128,1024]
    const float* W2    = (const float*)d_in[3];   // [128,512]
    const float* V     = (const float*)d_in[4];   // [1,128]

    float* out     = (float*)d_out;
    float* context = out;                 // 32*512
    float* weights = out + BATCH * KD;    // 32*2048

    char*   ws     = (char*)d_ws;
    float*  qproj  = (float*)ws;                                  //  16 KB
    float*  scores = (float*)(ws + 16384);                        // 256 KB
    __bf16* w2bf   = (__bf16*)(ws + 16384 + BATCH * SEQ * 4);     // 128 KB

    qproj_kernel  <<<dim3(BATCH),            dim3(HD),  0, stream>>>(query, W1, qproj);
    w2conv_kernel <<<dim3(128),              dim3(32),  0, stream>>>(W2, w2bf);
    scores_kernel <<<dim3(SEQ / 128, BATCH), dim3(256),
                     HD * KD * sizeof(__bf16), stream>>>(keys, qproj, V, w2bf, scores);
    softmax_kernel<<<dim3(BATCH),            dim3(256), 0, stream>>>(scores, weights);
    context_kernel<<<dim3(BATCH),            dim3(512), 0, stream>>>(keys, weights, context);
}
